// A3TGCNModel_11742440587921
// MI455X (gfx1250) — matrix operand order repaired and checked by reading, not compiled, not run on hardware
//
#include <hip/hip_runtime.h>
#include <math.h>

#define NN 50000
#define EE 1600000
#define CC 64
#define PP 10
#define NT16 (NN / 16)   // 3125 row tiles, exact

typedef __attribute__((ext_vector_type(16))) __bf16 v16bf;
typedef __attribute__((ext_vector_type(8)))  float  v8f;
typedef __attribute__((ext_vector_type(4)))  float  v4f;

// ---------- helpers ----------
__device__ __forceinline__ __bf16 f2bf(float f) {
    union { float f; unsigned u; } a; a.f = f;
    unsigned r = a.u + 0x7FFFu + ((a.u >> 16) & 1u);   // round-to-nearest-even
    union { unsigned short s; __bf16 b; } o; o.s = (unsigned short)(r >> 16);
    return o.b;
}
__device__ __forceinline__ float sigmoidf_(float x) { return 1.0f / (1.0f + __expf(-x)); }
__device__ __forceinline__ int bidx(int g, int kb, int nt, int lane) {
    return (((g * 2 + kb) * 4 + nt) * 32 + lane);
}
// Build a 16x32 bf16 A-fragment from f32 rows. base = &M[(row0+m)*CC]
// element i <-> K = (i>>3)*16 + h*8 + (i&7): two contiguous 8-float runs.
__device__ __forceinline__ v16bf load_a_frag(const float* __restrict__ base, int kb, int h) {
    const float* p0 = base + kb * 32 + h * 8;
    const float* p1 = p0 + 16;
    v4f f0 = *(const v4f*)p0, f1 = *(const v4f*)(p0 + 4);
    v4f f2 = *(const v4f*)p1, f3 = *(const v4f*)(p1 + 4);
    v16bf a;
#pragma unroll
    for (int i = 0; i < 4; i++) {
        a[i]      = f2bf(f0[i]);
        a[4 + i]  = f2bf(f1[i]);
        a[8 + i]  = f2bf(f2[i]);
        a[12 + i] = f2bf(f3[i]);
    }
    return a;
}

// ---------- prologue kernels ----------
__global__ void k_init_deg(float* deg) {
    int i = blockIdx.x * blockDim.x + threadIdx.x;
    if (i < NN) deg[i] = 1.0f;                       // self-loop
}
__global__ void k_deg_edges(const long long* __restrict__ ei, float* deg) {
    int e = blockIdx.x * blockDim.x + threadIdx.x;
    if (e < EE) atomicAdd(&deg[(int)ei[EE + e]], 1.0f);
}
__global__ void k_dinv(float* deg) {
    int i = blockIdx.x * blockDim.x + threadIdx.x;
    if (i < NN) { float d = deg[i]; deg[i] = d > 0.f ? rsqrtf(d) : 0.f; }
}
__global__ void k_init_S(const float* __restrict__ x, const float* __restrict__ dinv, float* S) {
    int i = blockIdx.x * blockDim.x + threadIdx.x;
    if (i < NN * PP) { int n = i / PP; float d = dinv[n]; S[i] = d * d * x[i]; }
}
__global__ void k_scatter(const long long* __restrict__ ei, const float* __restrict__ x,
                          const float* __restrict__ dinv, float* S) {
    int e = blockIdx.x * blockDim.x + threadIdx.x;
    if (e >= EE) return;
    int r = (int)ei[e], c = (int)ei[EE + e];
    float nm = dinv[r] * dinv[c];
#pragma unroll
    for (int t = 0; t < PP; t++) atomicAdd(&S[c * PP + t], nm * x[r * PP + t]);
}
__global__ void k_zero2(float* H, float* Hacc) {
    int i = blockIdx.x * blockDim.x + threadIdx.x;
    if (i < NN * CC) { H[i] = 0.f; Hacc[i] = 0.f; }
}
__global__ void k_probs(const float* __restrict__ att, float* probs) {
    if (threadIdx.x == 0 && blockIdx.x == 0) {
        float mx = att[0];
        for (int t = 1; t < PP; t++) mx = fmaxf(mx, att[t]);
        float e[PP], s = 0.f;
        for (int t = 0; t < PP; t++) { e[t] = __expf(att[t] - mx); s += e[t]; }
        for (int t = 0; t < PP; t++) probs[t] = e[t] / s;
    }
}
// u_g[j] = w_cg . w_lg[:C, j];  v_g[j] = b_cg . w_lg[:C, j] + b_lg[j]
__global__ void k_uv(const float* wcz, const float* bcz, const float* wcr, const float* bcr,
                     const float* wch, const float* bch,
                     const float* wlz, const float* blz, const float* wlr, const float* blr,
                     const float* wlh, const float* blh, float* u, float* v) {
    int tid = blockIdx.x * blockDim.x + threadIdx.x;
    if (tid >= 3 * CC) return;
    int g = tid >> 6, j = tid & 63;
    const float* wc = g == 0 ? wcz : (g == 1 ? wcr : wch);
    const float* bc = g == 0 ? bcz : (g == 1 ? bcr : bch);
    const float* wl = g == 0 ? wlz : (g == 1 ? wlr : wlh);
    const float* bl = g == 0 ? blz : (g == 1 ? blr : blh);
    float su = 0.f, sv = 0.f;
    for (int c = 0; c < CC; c++) { float w = wl[c * CC + j]; su += wc[c] * w; sv += bc[c] * w; }
    u[g * CC + j] = su;
    v[g * CC + j] = sv + bl[j];
}
// Pack Wh_g = w_lg[C:2C, :] into wave32 WMMA B-register layout:
// element i of lane: K = kb*32 + (i>>3)*16 + h*8 + (i&7), N = nt*16 + (lane&15)
__global__ void k_pack(const float* wlz, const float* wlr, const float* wlh, __bf16* Bp) {
    int tid = blockIdx.x * blockDim.x + threadIdx.x;
    if (tid >= 768) return;
    int lane = tid & 31, nt = (tid >> 5) & 3, kb = (tid >> 7) & 1, g = tid >> 8;
    const float* wl = g == 0 ? wlz : (g == 1 ? wlr : wlh);
    int h = lane >> 4, ncol = lane & 15;
    __bf16* dst = Bp + (size_t)bidx(g, kb, nt, lane) * 16;
#pragma unroll
    for (int i = 0; i < 16; i++) {
        int K = kb * 32 + (i >> 3) * 16 + h * 8 + (i & 7);
        dst[i] = f2bf(wl[(CC + K) * CC + nt * 16 + ncol]);
    }
}

// ---------- per-timestep GEMM kernels (WMMA) ----------
__global__ void __launch_bounds__(128)
k_gru_zr(const float* __restrict__ H, const float* __restrict__ S, const __bf16* __restrict__ Bp,
         const float* __restrict__ u, const float* __restrict__ vb,
         float* __restrict__ Z, float* __restrict__ HR, int t) {
    int wave = threadIdx.x >> 5, lane = threadIdx.x & 31;
    int rt = blockIdx.x * 4 + wave;
    if (rt >= NT16) return;
    int row0 = rt * 16, m = lane & 15, h = lane >> 4;
    const float* hrow = H + (size_t)(row0 + m) * CC;
    v16bf a0 = load_a_frag(hrow, 0, h);
    v16bf a1 = load_a_frag(hrow, 1, h);
    float sv[8];
#pragma unroll
    for (int vv = 0; vv < 8; vv++) sv[vv] = S[(row0 + vv + h * 8) * PP + t];
    const v16bf* BpV = (const v16bf*)Bp;
#pragma unroll
    for (int nt = 0; nt < 4; nt++) {
        v16bf bz0 = BpV[bidx(0, 0, nt, lane)], bz1 = BpV[bidx(0, 1, nt, lane)];
        v16bf br0 = BpV[bidx(1, 0, nt, lane)], br1 = BpV[bidx(1, 1, nt, lane)];
        v8f accz = {}, accr = {};
        accz = __builtin_amdgcn_wmma_f32_16x16x32_bf16(false, a0, false, bz0, (short)0, accz, false, false);
        accz = __builtin_amdgcn_wmma_f32_16x16x32_bf16(false, a1, false, bz1, (short)0, accz, false, false);
        accr = __builtin_amdgcn_wmma_f32_16x16x32_bf16(false, a0, false, br0, (short)0, accr, false, false);
        accr = __builtin_amdgcn_wmma_f32_16x16x32_bf16(false, a1, false, br1, (short)0, accr, false, false);
        int col = nt * 16 + m;
        float uz = u[col], ur = u[CC + col], vz = vb[col], vr = vb[CC + col];
#pragma unroll
        for (int vv = 0; vv < 8; vv++) {
            int n = row0 + vv + h * 8;
            size_t o = (size_t)n * CC + col;
            float z = sigmoidf_(sv[vv] * uz + accz[vv] + vz);
            float r = sigmoidf_(sv[vv] * ur + accr[vv] + vr);
            Z[o] = z;
            HR[o] = H[o] * r;
        }
    }
}

__global__ void __launch_bounds__(128)
k_gru_h(const float* __restrict__ HR, const float* __restrict__ S, const __bf16* __restrict__ Bp,
        const float* __restrict__ u, const float* __restrict__ vb, const float* __restrict__ Z,
        float* __restrict__ H, float* __restrict__ Hacc, const float* __restrict__ probs, int t) {
    int wave = threadIdx.x >> 5, lane = threadIdx.x & 31;
    int rt = blockIdx.x * 4 + wave;
    if (rt >= NT16) return;
    int row0 = rt * 16, m = lane & 15, h = lane >> 4;
    const float* hrow = HR + (size_t)(row0 + m) * CC;
    v16bf a0 = load_a_frag(hrow, 0, h);
    v16bf a1 = load_a_frag(hrow, 1, h);
    float sv[8];
#pragma unroll
    for (int vv = 0; vv < 8; vv++) sv[vv] = S[(row0 + vv + h * 8) * PP + t];
    float pt = probs[t];
    const v16bf* BpV = (const v16bf*)Bp;
#pragma unroll
    for (int nt = 0; nt < 4; nt++) {
        v16bf b0 = BpV[bidx(2, 0, nt, lane)], b1 = BpV[bidx(2, 1, nt, lane)];
        v8f acc = {};
        acc = __builtin_amdgcn_wmma_f32_16x16x32_bf16(false, a0, false, b0, (short)0, acc, false, false);
        acc = __builtin_amdgcn_wmma_f32_16x16x32_bf16(false, a1, false, b1, (short)0, acc, false, false);
        int col = nt * 16 + m;
        float uh = u[2 * CC + col], vh = vb[2 * CC + col];
#pragma unroll
        for (int vv = 0; vv < 8; vv++) {
            int n = row0 + vv + h * 8;
            size_t o = (size_t)n * CC + col;
            float ht = tanhf(sv[vv] * uh + acc[vv] + vh);
            float z = Z[o];
            float hn = z * H[o] + (1.f - z) * ht;
            H[o] = hn;
            Hacc[o] += pt * hn;
        }
    }
}

// ---------- output head ----------
__global__ void k_out(const float* __restrict__ Hacc, const float* __restrict__ wout,
                      const float* __restrict__ bout, float* __restrict__ out) {
    int n = blockIdx.x * blockDim.x + threadIdx.x;
    if (n >= NN) return;
    const float* hp = Hacc + (size_t)n * CC;
    float s = bout[0];
#pragma unroll
    for (int c = 0; c < CC; c++) s += fmaxf(hp[c], 0.f) * wout[c];
    out[n] = s;
}

// ---------- workspace layout (bytes, 256-aligned) ----------
#define OFF_DINV  0ull
#define OFF_S     200704ull
#define OFF_H     2200832ull
#define OFF_HACC  15000832ull
#define OFF_HR    27800832ull
#define OFF_Z     40600832ull
#define OFF_U     53400832ull
#define OFF_V     53401600ull
#define OFF_PROBS 53402368ull
#define OFF_BP    53402624ull

extern "C" void kernel_launch(void* const* d_in, const int* in_sizes, int n_in,
                              void* d_out, int out_size, void* d_ws, size_t ws_size,
                              hipStream_t stream) {
    const float*     x   = (const float*)d_in[0];
    const long long* ei  = (const long long*)d_in[1];
    const float *wcz = (const float*)d_in[2],  *bcz = (const float*)d_in[3];
    const float *wcr = (const float*)d_in[4],  *bcr = (const float*)d_in[5];
    const float *wch = (const float*)d_in[6],  *bch = (const float*)d_in[7];
    const float *wlz = (const float*)d_in[8],  *blz = (const float*)d_in[9];
    const float *wlr = (const float*)d_in[10], *blr = (const float*)d_in[11];
    const float *wlh = (const float*)d_in[12], *blh = (const float*)d_in[13];
    const float *att = (const float*)d_in[14];
    const float *wout = (const float*)d_in[15], *bout = (const float*)d_in[16];

    char* ws = (char*)d_ws;
    float*  dinv  = (float*)(ws + OFF_DINV);
    float*  S     = (float*)(ws + OFF_S);
    float*  H     = (float*)(ws + OFF_H);
    float*  Hacc  = (float*)(ws + OFF_HACC);
    float*  HR    = (float*)(ws + OFF_HR);
    float*  Z     = (float*)(ws + OFF_Z);
    float*  u     = (float*)(ws + OFF_U);
    float*  v     = (float*)(ws + OFF_V);
    float*  probs = (float*)(ws + OFF_PROBS);
    __bf16* Bp    = (__bf16*)(ws + OFF_BP);

    // graph normalization + scalar SpMV (all P periods at once)
    k_init_deg <<<(NN + 255) / 256, 256, 0, stream>>>(dinv);
    k_deg_edges<<<(EE + 255) / 256, 256, 0, stream>>>(ei, dinv);
    k_dinv     <<<(NN + 255) / 256, 256, 0, stream>>>(dinv);
    k_init_S   <<<(NN * PP + 255) / 256, 256, 0, stream>>>(x, dinv, S);
    k_scatter  <<<(EE + 255) / 256, 256, 0, stream>>>(ei, x, dinv, S);

    // state + folded weights
    k_zero2<<<(NN * CC + 255) / 256, 256, 0, stream>>>(H, Hacc);
    k_probs<<<1, 32, 0, stream>>>(att, probs);
    k_uv   <<<1, 192, 0, stream>>>(wcz, bcz, wcr, bcr, wch, bch,
                                   wlz, blz, wlr, blr, wlh, blh, u, v);
    k_pack <<<3, 256, 0, stream>>>(wlz, wlr, wlh, Bp);

    // sequential GRU over P timesteps (WMMA GEMMs)
    int gblk = (NT16 + 3) / 4;
    for (int t = 0; t < PP; t++) {
        k_gru_zr<<<gblk, 128, 0, stream>>>(H, S, Bp, u, v, Z, HR, t);
        k_gru_h <<<gblk, 128, 0, stream>>>(HR, S, Bp, u, v, Z, H, Hacc, probs, t);
    }

    // ReLU + Linear(C->1)
    k_out<<<(NN + 255) / 256, 256, 0, stream>>>(Hacc, wout, bout, (float*)d_out);
}
